// FactorizedL2VQ_22256520528537
// MI455X (gfx1250) — matrix operand order repaired and checked by reading, not compiled, not run on hardware
//
#include <hip/hip_runtime.h>
#include <hip/hip_bf16.h>

typedef __attribute__((ext_vector_type(16))) _Float16 v16h;
typedef __attribute__((ext_vector_type(8)))  _Float16 v8h;
typedef __attribute__((ext_vector_type(8)))  float    v8f;

#define NB   16
#define NC   256
#define NHW  4096          // 64*64
#define NTOK 65536         // NB*NHW
#define ND   16
#define NK   4096
#define DECAY 0.99f
#define BETA  0.25f

// ---------------------------------------------------------------- zero ws
__global__ void vq_zero(float* p, int n) {
    int i = blockIdx.x * 256 + threadIdx.x;
    if (i < n) p[i] = 0.0f;
}

// ------------------------------------------------- normalize embedding rows
__global__ void vq_norm_embed(const float* __restrict__ emb,
                              float* __restrict__ e_f32,
                              _Float16* __restrict__ e_h) {
    int k = blockIdx.x * 256 + threadIdx.x;
    if (k >= NK) return;
    float v[ND]; float ss = 0.0f;
#pragma unroll
    for (int d = 0; d < ND; ++d) { v[d] = emb[k * ND + d]; ss += v[d] * v[d]; }
    float inv = 1.0f / fmaxf(sqrtf(ss), 1e-12f);
#pragma unroll
    for (int d = 0; d < ND; ++d) {
        float nv = v[d] * inv;
        e_f32[k * ND + d] = nv;
        e_h[k * ND + d]   = (_Float16)nv;
    }
}

// ---------------------------------------- down-projection + L2 normalize
__global__ void __launch_bounds__(256)
vq_down(const float* __restrict__ z, const float* __restrict__ w_down,
        const float* __restrict__ b_down,
        float* __restrict__ zn_f32, _Float16* __restrict__ zn_h) {
    __shared__ __align__(16) float wlds[NC * ND];   // transposed: [c][d]
    int tid = threadIdx.x;
    for (int i = tid; i < NC * ND; i += 256) {
        int d = i >> 8, c = i & 255;                // i = d*256+c
        wlds[c * ND + d] = w_down[i];
    }
    __syncthreads();

    int n = blockIdx.x * 256 + tid;
    int b = n >> 12, p = n & (NHW - 1);
    const float* zp = z + (size_t)b * NC * NHW + p;

    float acc[ND];
#pragma unroll
    for (int d = 0; d < ND; ++d) acc[d] = 0.0f;

    for (int c = 0; c < NC; ++c) {
        float zv = zp[(size_t)c * NHW];
        const float4* wv = (const float4*)(wlds + c * ND);
        float4 w0 = wv[0], w1 = wv[1], w2 = wv[2], w3 = wv[3];
        acc[0]  += w0.x * zv; acc[1]  += w0.y * zv; acc[2]  += w0.z * zv; acc[3]  += w0.w * zv;
        acc[4]  += w1.x * zv; acc[5]  += w1.y * zv; acc[6]  += w1.z * zv; acc[7]  += w1.w * zv;
        acc[8]  += w2.x * zv; acc[9]  += w2.y * zv; acc[10] += w2.z * zv; acc[11] += w2.w * zv;
        acc[12] += w3.x * zv; acc[13] += w3.y * zv; acc[14] += w3.z * zv; acc[15] += w3.w * zv;
    }
    float ss = 0.0f;
#pragma unroll
    for (int d = 0; d < ND; ++d) { acc[d] += b_down[d]; ss += acc[d] * acc[d]; }
    float inv = 1.0f / fmaxf(sqrtf(ss), 1e-12f);
#pragma unroll
    for (int d = 0; d < ND; ++d) {
        float nv = acc[d] * inv;
        zn_f32[(size_t)n * ND + d] = nv;
        zn_h[(size_t)n * ND + d]   = (_Float16)nv;
    }
}

// -------------------------------------------- WMMA argmax over 4096 codes
// each wave: 32 tokens (2 A tiles); block: 8 waves = 256 tokens.
// e_norm (f16) staged through LDS in 32KB chunks; 8 WMMAs per iteration.
__global__ void __launch_bounds__(256)
vq_argmax(const _Float16* __restrict__ z_h, const _Float16* __restrict__ e_h,
          int* __restrict__ idx_ws, float* __restrict__ idx_out) {
    __shared__ __align__(32) _Float16 elds[1024 * ND];   // 32 KB chunk of codes
    __shared__ __align__(32) _Float16 zbuf[ND];          // 32 B of zeros
    const int tid  = threadIdx.x;
    const int lane = tid & 31;
    const int wave = tid >> 5;
    const int tokenBase = (blockIdx.x * 8 + wave) * 32;

    if (tid < ND) zbuf[tid] = (_Float16)0.0f;

    // A tiles: 16 tokens x K32 f16 each (K=0..15 real, 16..31 zero)
    const int M  = lane & 15;
    v16h a0 = {}, a1 = {};
    if (lane < 16) {
        v8h lo0 = *(const v8h*)(z_h + ((size_t)(tokenBase + M) * ND));
        v8h hi0 = *(const v8h*)(z_h + ((size_t)(tokenBase + M) * ND + 8));
        v8h lo1 = *(const v8h*)(z_h + ((size_t)(tokenBase + 16 + M) * ND));
        v8h hi1 = *(const v8h*)(z_h + ((size_t)(tokenBase + 16 + M) * ND + 8));
#pragma unroll
        for (int j = 0; j < 8; ++j) { a0[j] = lo0[j]; a0[8 + j] = hi0[j]; a1[j] = lo1[j]; a1[8 + j] = hi1[j]; }
    }

    float mx0[8], mx1[8]; int bi0[8], bi1[8];
#pragma unroll
    for (int r = 0; r < 8; ++r) {
        mx0[r] = -3.0e38f; mx1[r] = -3.0e38f; bi0[r] = 0; bi1[r] = 0;
    }
    v8f cz = {};

    // Branch-free B source: lanes 0..15 walk code rows (stride 16 rows per
    // step), lanes 16..31 always re-read the 32B zero line (stride 0).
    const _Float16* bbase = (lane < 16) ? (elds + lane * ND) : zbuf;
    const int bstep = (lane < 16) ? (16 * ND) : 0;   // halves per c16 step

    for (int co = 0; co < NK / 1024; ++co) {
        __syncthreads();
        const uint4* src = (const uint4*)(e_h + (size_t)co * 1024 * ND);
        uint4* dst = (uint4*)elds;
        for (int i = tid; i < 1024 * ND * 2 / 16; i += 256) dst[i] = src[i];
        __syncthreads();

        const _Float16* bp = bbase;
        int code0 = co * 1024 + (lane & 15);
        for (int c16 = 0; c16 < 64; c16 += 4) {
            v16h b0 = *(const v16h*)bp;
            v16h b1 = *(const v16h*)(bp + bstep);
            v16h b2 = *(const v16h*)(bp + 2 * bstep);
            v16h b3 = *(const v16h*)(bp + 3 * bstep);
            bp += 4 * bstep;
            // 8 independent WMMAs: one dscnt wait + hazard window per 8 ops;
            // compares dual-issue/co-execute alongside the XDL pipe.
            v8f s00 = __builtin_amdgcn_wmma_f32_16x16x32_f16(
                false, a0, false, b0, (short)0, cz, false, false);
            v8f s10 = __builtin_amdgcn_wmma_f32_16x16x32_f16(
                false, a1, false, b0, (short)0, cz, false, false);
            v8f s01 = __builtin_amdgcn_wmma_f32_16x16x32_f16(
                false, a0, false, b1, (short)0, cz, false, false);
            v8f s11 = __builtin_amdgcn_wmma_f32_16x16x32_f16(
                false, a1, false, b1, (short)0, cz, false, false);
            v8f s02 = __builtin_amdgcn_wmma_f32_16x16x32_f16(
                false, a0, false, b2, (short)0, cz, false, false);
            v8f s12 = __builtin_amdgcn_wmma_f32_16x16x32_f16(
                false, a1, false, b2, (short)0, cz, false, false);
            v8f s03 = __builtin_amdgcn_wmma_f32_16x16x32_f16(
                false, a0, false, b3, (short)0, cz, false, false);
            v8f s13 = __builtin_amdgcn_wmma_f32_16x16x32_f16(
                false, a1, false, b3, (short)0, cz, false, false);
            int code1 = code0 + 16;
            int code2 = code0 + 32;
            int code3 = code0 + 48;
#pragma unroll
            for (int r = 0; r < 8; ++r) {
                if (s00[r] > mx0[r]) { mx0[r] = s00[r]; bi0[r] = code0; }
                if (s10[r] > mx1[r]) { mx1[r] = s10[r]; bi1[r] = code0; }
                if (s01[r] > mx0[r]) { mx0[r] = s01[r]; bi0[r] = code1; }
                if (s11[r] > mx1[r]) { mx1[r] = s11[r]; bi1[r] = code1; }
                if (s02[r] > mx0[r]) { mx0[r] = s02[r]; bi0[r] = code2; }
                if (s12[r] > mx1[r]) { mx1[r] = s12[r]; bi1[r] = code2; }
                if (s03[r] > mx0[r]) { mx0[r] = s03[r]; bi0[r] = code3; }
                if (s13[r] > mx1[r]) { mx1[r] = s13[r]; bi1[r] = code3; }
            }
            code0 += 64;
        }
    }

    // butterfly argmax across the 16 lanes of each wave half
#pragma unroll
    for (int t = 0; t < 2; ++t) {
#pragma unroll
        for (int r = 0; r < 8; ++r) {
            float mv = t ? mx1[r] : mx0[r];
            int   iv = t ? bi1[r] : bi0[r];
#pragma unroll
            for (int off = 8; off >= 1; off >>= 1) {
                float om = __shfl_xor(mv, off, 32);
                int   oi = __shfl_xor(iv, off, 32);
                if (om > mv || (om == mv && oi < iv)) { mv = om; iv = oi; }
            }
            if ((lane & 15) == 0) {
                int token = tokenBase + t * 16 + r + ((lane < 16) ? 0 : 8);
                idx_ws[token]  = iv;
                idx_out[token] = (float)iv;
            }
        }
    }
}

// ----------------------------------------------- gather + up-projection
__global__ void __launch_bounds__(256)
vq_up(const float* __restrict__ e_f32, const int* __restrict__ idx,
      const float* __restrict__ w_up, const float* __restrict__ b_up,
      float* __restrict__ zq) {
    __shared__ __align__(16) float wlds[NC * ND];
    __shared__ float blds[NC];
    int tid = threadIdx.x;
    for (int i = tid; i < NC * ND; i += 256) wlds[i] = w_up[i];
    if (tid < NC) blds[tid] = b_up[tid];
    __syncthreads();

    int n = blockIdx.x * 256 + tid;
    int b = n >> 12, p = n & (NHW - 1);
    int k = idx[n];
    float q[ND];
#pragma unroll
    for (int d = 0; d < ND; ++d) q[d] = e_f32[k * ND + d];

    float* op = zq + (size_t)b * NC * NHW + p;
    for (int c = 0; c < NC; ++c) {
        const float4* wv = (const float4*)(wlds + c * ND);
        float4 w0 = wv[0], w1 = wv[1], w2 = wv[2], w3 = wv[3];
        float s = blds[c];
        s += q[0]*w0.x + q[1]*w0.y + q[2]*w0.z + q[3]*w0.w;
        s += q[4]*w1.x + q[5]*w1.y + q[6]*w1.z + q[7]*w1.w;
        s += q[8]*w2.x + q[9]*w2.y + q[10]*w2.z + q[11]*w2.w;
        s += q[12]*w3.x + q[13]*w3.y + q[14]*w3.z + q[15]*w3.w;
        op[(size_t)c * NHW] = s;
    }
}

// --------------------------- scatter: bincount, embed_sum, commitment sum
__global__ void __launch_bounds__(256)
vq_scatter(const int* __restrict__ idx, const float* __restrict__ zn,
           const float* __restrict__ e_f32,
           float* __restrict__ bincount, float* __restrict__ esum,
           float* __restrict__ scalars) {
    __shared__ float red[256];
    int tid = threadIdx.x;
    int n = blockIdx.x * 256 + tid;
    int k = idx[n];
    atomicAdd(&bincount[k], 1.0f);
    float sq = 0.0f;
#pragma unroll
    for (int d = 0; d < ND; ++d) {
        float zv = zn[(size_t)n * ND + d];
        atomicAdd(&esum[k * ND + d], zv);
        float df = zv - e_f32[k * ND + d];
        sq += df * df;
    }
    red[tid] = sq;
    __syncthreads();
    for (int s = 128; s > 0; s >>= 1) {
        if (tid < s) red[tid] += red[tid + s];
        __syncthreads();
    }
    if (tid == 0) atomicAdd(&scalars[2], red[0]);
}

// ------------------------------------ EMA update per code + entropy/n sums
__global__ void vq_ema(const float* __restrict__ cluster_size,
                       const float* __restrict__ embed_avg,
                       const float* __restrict__ bincount,
                       const float* __restrict__ esum,
                       float* __restrict__ out_ncs, float* __restrict__ out_nea,
                       float* __restrict__ scalars) {
    int k = blockIdx.x * 256 + threadIdx.x;
    if (k >= NK) return;
    float bc  = bincount[k];
    float ncs = cluster_size[k] * DECAY + (1.0f - DECAY) * bc;
    out_ncs[k] = ncs;
    atomicAdd(&scalars[0], ncs);
    float p = bc / ((float)NTOK + 1e-8f);
    atomicAdd(&scalars[1], -p * logf(p + 1e-8f));
#pragma unroll
    for (int d = 0; d < ND; ++d)
        out_nea[k * ND + d] =
            embed_avg[k * ND + d] * DECAY + (1.0f - DECAY) * esum[k * ND + d];
}

// -------------------------------- finalize: new_embedding + scalar outputs
__global__ void vq_final(const float* __restrict__ out_ncs,
                         const float* __restrict__ out_nea,
                         const float* __restrict__ scalars,
                         float* __restrict__ out_ne,
                         float* __restrict__ out_commit,
                         float* __restrict__ out_entropy) {
    int t = blockIdx.x * 256 + threadIdx.x;
    if (t >= NK * ND) return;
    int k = t >> 4;
    float nv  = scalars[0];
    float cs  = (out_ncs[k] + 1e-5f) / (nv + (float)NK * 1e-5f) * nv;
    out_ne[t] = out_nea[t] / cs;
    if (t == 0) {
        out_commit[0]  = BETA * scalars[2] / ((float)NTOK * (float)ND);
        out_entropy[0] = logf((float)NK) - scalars[1];
    }
}

extern "C" void kernel_launch(void* const* d_in, const int* in_sizes, int n_in,
                              void* d_out, int out_size, void* d_ws, size_t ws_size,
                              hipStream_t stream) {
    const float* z            = (const float*)d_in[0];
    const float* w_down       = (const float*)d_in[1];
    const float* b_down       = (const float*)d_in[2];
    const float* w_up         = (const float*)d_in[3];
    const float* b_up         = (const float*)d_in[4];
    const float* embedding    = (const float*)d_in[5];
    const float* cluster_size = (const float*)d_in[6];
    const float* embed_avg    = (const float*)d_in[7];

    // ---- workspace layout ----
    float*     e_f32    = (float*)d_ws;                    // 65536
    float*     bincount = e_f32 + NK * ND;                 // 4096
    float*     esum     = bincount + NK;                   // 65536
    float*     scalars  = esum + NK * ND;                  // 4: n_sum, ent_sum, commit_sum
    float*     zn_f32   = scalars + 4;                     // 1048576
    _Float16*  e_h      = (_Float16*)(zn_f32 + (size_t)NTOK * ND);   // 65536 halves
    _Float16*  zn_h     = e_h + NK * ND;                   // 1048576 halves
    int*       idx_ws   = (int*)(zn_h + (size_t)NTOK * ND);          // 65536 ints

    // ---- output layout (flat concat, return order) ----
    float* out         = (float*)d_out;
    float* out_zq      = out;                              // 16777216
    float* out_commit  = out + 16777216;                   // 1
    float* out_idx     = out + 16777217;                   // 65536
    float* out_entropy = out + 16842753;                   // 1
    float* out_ne      = out + 16842754;                   // 65536
    float* out_ncs     = out + 16908290;                   // 4096
    float* out_nea     = out + 16912386;                   // 65536

    const int zero_n = NK + NK * ND + 4;                   // bincount..scalars
    vq_zero<<<(zero_n + 255) / 256, 256, 0, stream>>>(bincount, zero_n);
    vq_norm_embed<<<NK / 256, 256, 0, stream>>>(embedding, e_f32, e_h);
    vq_down<<<NTOK / 256, 256, 0, stream>>>(z, w_down, b_down, zn_f32, zn_h);
    vq_argmax<<<NTOK / 256, 256, 0, stream>>>(zn_h, e_h, idx_ws, out_idx);
    vq_up<<<NTOK / 256, 256, 0, stream>>>(e_f32, idx_ws, w_up, b_up, out_zq);
    vq_scatter<<<NTOK / 256, 256, 0, stream>>>(idx_ws, zn_f32, e_f32,
                                               bincount, esum, scalars);
    vq_ema<<<NK / 256, 256, 0, stream>>>(cluster_size, embed_avg, bincount,
                                         esum, out_ncs, out_nea, scalars);
    vq_final<<<(NK * ND) / 256, 256, 0, stream>>>(out_ncs, out_nea, scalars,
                                                  out_ne, out_commit, out_entropy);
}
